// QuantizedViTAttention_32109175505452
// MI455X (gfx1250) — compile-verified, matrix operand order checked by reading, not run on hardware
//
#include <hip/hip_runtime.h>
#include <hip/hip_bf16.h>
#include <math.h>

// ---------------------------------------------------------------------------
// QuantizedViTAttention for MI455X (gfx1250, wave32, WMMA)
//   B=16, S=577, D=1024, H=16, hd=64
//   All matrix math via v_wmma_f32_16x16x32_f16; every fragment is two
//   16-byte loads per lane; guards are index-clamps (no EXEC branching).
// ---------------------------------------------------------------------------

#define Bdim 16
#define Sdim 577
#define Ddim 1024
#define Hdim 16
#define HD   64
#define NTOK (Bdim * Sdim)   // 9232
#define SPAD 608             // 19 key-chunks * 32, keeps V rows 16B-aligned

typedef __attribute__((ext_vector_type(16))) _Float16 v16h;
typedef __attribute__((ext_vector_type(8)))  _Float16 v8h;
typedef __attribute__((ext_vector_type(8)))  float    v8f;

union AF {
    v16h v;
    v8h  p[2];
    _Float16 h[16];
};

// ---------------------------------------------------------------------------
// Kernel 1: symmetric per-row 8-bit fake quant (quantize->dequantize),
// stored as f16 in the ORIGINAL [o][d] orientation (d = contraction dim
// innermost => WMMA B fragments are two contiguous 16B loads per lane).
// jnp.round is round-half-to-even -> rintf (default RNE mode).
// ---------------------------------------------------------------------------
__global__ void __launch_bounds__(256) quant_w(const float* __restrict__ Wq,
                                               const float* __restrict__ Wk,
                                               const float* __restrict__ Wv,
                                               _Float16* __restrict__ wdq) {
    const int o  = blockIdx.x;      // output channel (row of W)
    const int wi = blockIdx.y;      // 0=Q 1=K 2=V
    const float* W = (wi == 0) ? Wq : (wi == 1) ? Wk : Wv;
    const float* row = W + (size_t)o * Ddim;

    __shared__ float red[256];
    float m = 0.f;
    for (int d = threadIdx.x; d < Ddim; d += 256) m = fmaxf(m, fabsf(row[d]));
    red[threadIdx.x] = m;
    __syncthreads();
    for (int s = 128; s > 0; s >>= 1) {
        if (threadIdx.x < s) red[threadIdx.x] = fmaxf(red[threadIdx.x], red[threadIdx.x + s]);
        __syncthreads();
    }
    const float scale = fmaxf(red[0] * (1.0f / 127.0f), 1e-8f);
    const float inv   = 1.0f / scale;

    _Float16* out = wdq + (size_t)wi * Ddim * Ddim + (size_t)o * Ddim;
    for (int d = threadIdx.x; d < Ddim; d += 256) {
        float q = rintf(row[d] * inv);
        q = fminf(fmaxf(q, -127.f), 127.f);
        out[d] = (_Float16)(q * scale);      // coalesced
    }
}

// ---------------------------------------------------------------------------
// Kernel 2: y = x @ Wdq^T + b.
//   Q,K written head-split [b][h][s][64] (f16).
//   V written TRANSPOSED   [b][h][64][SPAD] (f16) for contiguous PV fragments.
// Block tile 128(M) x 64(N), 8 waves, each wave -> 16x64, K-steps of 32.
// A staged via LDS (f32->f16, b128 in/out); B loaded straight from global.
// Row guards are index-clamps; only the epilogue stores are predicated.
// ---------------------------------------------------------------------------
__global__ void __launch_bounds__(256) qkv_gemm(const float*    __restrict__ x,
                                                const _Float16* __restrict__ wdq,
                                                const float*    __restrict__ bq,
                                                const float*    __restrict__ bk,
                                                const float*    __restrict__ bv,
                                                _Float16*       __restrict__ q_out,
                                                _Float16*       __restrict__ k_out,
                                                _Float16*       __restrict__ v_out) {
    const int wi = blockIdx.z;
    const float*    bias = (wi == 0) ? bq : (wi == 1) ? bk : bv;
    const _Float16* wf   = wdq + (size_t)wi * Ddim * Ddim;   // [o][d]

    const int m0 = blockIdx.x * 128;
    const int n0 = blockIdx.y * 64;

    __shared__ _Float16 xs[128][32 + 8];   // rows are 80B => 16B-aligned slots

    const int tid  = threadIdx.x;
    const int lane = tid & 31;
    const int wv   = tid >> 5;
    const int g    = lane >> 4;
    const int ln   = lane & 15;

    // staging thread's (clamped) source row: OOB rows read duplicate data,
    // they are never stored.
    const int sr  = tid >> 1;
    const int sc_ = (tid & 1) * 16;
    const int gm  = min(m0 + sr, NTOK - 1);
    const float4* s4base = (const float4*)(x + (size_t)gm * Ddim + sc_);

    v8f acc[4] = {};

    for (int k0 = 0; k0 < Ddim; k0 += 32) {
        // ---- stage x tile 128x32 (f32 -> f16), unconditional b128 loads --
        {
            const float4* s4 = s4base + (k0 >> 2);
            _Float16 tmp[16];
#pragma unroll
            for (int j = 0; j < 4; ++j) {
                const float4 f = s4[j];
                tmp[j * 4 + 0] = (_Float16)f.x;
                tmp[j * 4 + 1] = (_Float16)f.y;
                tmp[j * 4 + 2] = (_Float16)f.z;
                tmp[j * 4 + 3] = (_Float16)f.w;
            }
            *(v8h*)&xs[sr][sc_]     = *(v8h*)&tmp[0];
            *(v8h*)&xs[sr][sc_ + 8] = *(v8h*)&tmp[8];
        }
        __syncthreads();

        // ---- A fragment: two ds_load_b128 --------------------------------
        AF a;
        a.p[0] = *(const v8h*)&xs[wv * 16 + ln][g * 8];
        a.p[1] = *(const v8h*)&xs[wv * 16 + ln][16 + g * 8];

        // ---- B fragments straight from global (L2-resident weights) ------
#pragma unroll
        for (int nt = 0; nt < 4; ++nt) {
            const int o = n0 + nt * 16 + ln;
            const _Float16* wr = wf + (size_t)o * Ddim + k0;
            AF bfrag;
            bfrag.p[0] = *(const v8h*)(wr + g * 8);
            bfrag.p[1] = *(const v8h*)(wr + 16 + g * 8);
            acc[nt] = __builtin_amdgcn_wmma_f32_16x16x32_f16(
                false, a.v, false, bfrag.v, (short)0, acc[nt], false, false);
        }
        __syncthreads();
    }

    // ---- epilogue: +bias, store f16 --------------------------------------
#pragma unroll
    for (int nt = 0; nt < 4; ++nt) {
        const int   o  = n0 + nt * 16 + ln;
        const float bb = bias[o];
        const int   hh = o >> 6;
        const int   dd = o & 63;
#pragma unroll
        for (int r = 0; r < 8; ++r) {
            const int mrow = m0 + wv * 16 + r + (g << 3);
            if (mrow < NTOK) {
                const int b = mrow / Sdim;
                const int s = mrow - b * Sdim;
                const _Float16 val = (_Float16)(acc[nt][r] + bb);
                if (wi == 2) {
                    // V transposed: [b][h][dd][SPAD]
                    v_out[(((size_t)b * Hdim + hh) * HD + dd) * SPAD + s] = val;
                } else {
                    _Float16* out = (wi == 0) ? q_out : k_out;
                    out[(((size_t)b * Hdim + hh) * Sdim + s) * HD + dd] = val;
                }
            }
        }
    }
}

// ---------------------------------------------------------------------------
// Kernel 3: flash attention. One wave owns a 16-query tile; key chunks of 32.
// scores = QK^T / 8 (4 WMMAs/chunk), online softmax, ctx += P V (4 WMMAs).
// V fragments are prefetched before the softmax so exp/shuffle VALU work
// overlaps the global load latency. Guards are index-clamps; masked keys are
// forced to -inf AFTER the matmul, so clamped loads are harmless.
// ---------------------------------------------------------------------------
__global__ void __launch_bounds__(256) attn(const _Float16* __restrict__ qbuf,
                                            const _Float16* __restrict__ kbuf,
                                            const _Float16* __restrict__ vbuf,
                                            float*          __restrict__ outp) {
    const int bh   = blockIdx.x;           // b*H + h
    const int qblk = blockIdx.y;           // 128 queries per block
    const int tid  = threadIdx.x;
    const int lane = tid & 31;
    const int wv   = tid >> 5;
    const int g    = lane >> 4;
    const int ln   = lane & 15;
    const int q0   = qblk * 128 + wv * 16;

    const _Float16* Qb = qbuf + (size_t)bh * Sdim * HD;
    const _Float16* Kb = kbuf + (size_t)bh * Sdim * HD;
    const _Float16* Vb = vbuf + (size_t)bh * HD * SPAD;   // [d][SPAD]

    __shared__ _Float16 pls[8][16][32 + 8];   // per-wave, rows 16B-aligned

    // ---- Q fragments: clamped row, unconditional b128 loads --------------
    AF qa[2];
    {
        const int row = min(q0 + ln, Sdim - 1);
#pragma unroll
        for (int c = 0; c < 2; ++c) {
            const _Float16* qr = Qb + (size_t)row * HD + c * 32;
            qa[c].p[0] = *(const v8h*)(qr + g * 8);
            qa[c].p[1] = *(const v8h*)(qr + 16 + g * 8);
        }
    }

    float mrow[8], lrow[8];
#pragma unroll
    for (int r = 0; r < 8; ++r) { mrow[r] = -3.0e38f; lrow[r] = 0.f; }
    v8f acc[4] = {};
    const float sm_scale = 0.125f;  // 1/sqrt(64)

    for (int kc = 0; kc < Sdim; kc += 32) {
        // ---- scores: two 16x16 tiles covering 32 keys --------------------
        v8f sc[2] = {};
#pragma unroll
        for (int t = 0; t < 2; ++t) {
            const int key = min(kc + t * 16 + ln, Sdim - 1);   // clamped load
            const _Float16* kr0 = Kb + (size_t)key * HD;
#pragma unroll
            for (int c = 0; c < 2; ++c) {
                AF bf;  // K^T fragment: (k=d, n=key) — contiguous in d
                const _Float16* kr = kr0 + c * 32;
                bf.p[0] = *(const v8h*)(kr + g * 8);
                bf.p[1] = *(const v8h*)(kr + 16 + g * 8);
                sc[t] = __builtin_amdgcn_wmma_f32_16x16x32_f16(
                    false, qa[c].v, false, bf.v, (short)0, sc[t], false, false);
            }
        }

        // ---- prefetch V fragments (independent of softmax) ---------------
        AF vf[4];
#pragma unroll
        for (int nt = 0; nt < 4; ++nt) {
            const _Float16* vr = Vb + (size_t)(nt * 16 + ln) * SPAD + kc;
            vf[nt].p[0] = *(const v8h*)(vr + g * 8);
            vf[nt].p[1] = *(const v8h*)(vr + 16 + g * 8);
        }

        // ---- mask + scale, row max (reduce inside 16-lane half-wave) -----
        float newm[8];
#pragma unroll
        for (int r = 0; r < 8; ++r) {
            const float v0 = (kc + ln      < Sdim) ? sc[0][r] * sm_scale : -3.0e38f;
            const float v1 = (kc + 16 + ln < Sdim) ? sc[1][r] * sm_scale : -3.0e38f;
            sc[0][r] = v0; sc[1][r] = v1;
            newm[r]  = fmaxf(v0, v1);
        }
#pragma unroll
        for (int m = 1; m < 16; m <<= 1)
#pragma unroll
            for (int r = 0; r < 8; ++r)
                newm[r] = fmaxf(newm[r], __shfl_xor(newm[r], m, 32));

        float rescale[8], psum[8];
#pragma unroll
        for (int r = 0; r < 8; ++r) {
            newm[r]    = fmaxf(newm[r], mrow[r]);
            rescale[r] = __expf(mrow[r] - newm[r]);
            mrow[r]    = newm[r];
            const float p0 = __expf(sc[0][r] - newm[r]);
            const float p1 = __expf(sc[1][r] - newm[r]);
            sc[0][r] = p0; sc[1][r] = p1;
            psum[r]  = p0 + p1;
        }
#pragma unroll
        for (int m = 1; m < 16; m <<= 1)
#pragma unroll
            for (int r = 0; r < 8; ++r)
                psum[r] += __shfl_xor(psum[r], m, 32);
#pragma unroll
        for (int r = 0; r < 8; ++r) lrow[r] = lrow[r] * rescale[r] + psum[r];

        // rescale running context accumulators (same C row layout)
#pragma unroll
        for (int nt = 0; nt < 4; ++nt)
#pragma unroll
            for (int r = 0; r < 8; ++r) acc[nt][r] *= rescale[r];

        // ---- P: C-layout scatter to per-wave LDS, A-layout b128 reads ----
#pragma unroll
        for (int t = 0; t < 2; ++t)
#pragma unroll
            for (int r = 0; r < 8; ++r)
                pls[wv][r + (g << 3)][t * 16 + ln] = (_Float16)sc[t][r];

        AF pa;
        pa.p[0] = *(const v8h*)&pls[wv][ln][g * 8];
        pa.p[1] = *(const v8h*)&pls[wv][ln][16 + g * 8];

        // ---- ctx += P @ V : V transposed => contiguous key runs ----------
#pragma unroll
        for (int nt = 0; nt < 4; ++nt) {
            acc[nt] = __builtin_amdgcn_wmma_f32_16x16x32_f16(
                false, pa.v, false, vf[nt].v, (short)0, acc[nt], false, false);
        }
    }

    // ---- epilogue: divide by row sum, store fp32 [B,S,D] -----------------
    const int b  = bh >> 4;
    const int hh = bh & 15;
#pragma unroll
    for (int nt = 0; nt < 4; ++nt)
#pragma unroll
        for (int r = 0; r < 8; ++r) {
            const int row = q0 + r + (g << 3);
            if (row < Sdim) {
                const float val = acc[nt][r] / lrow[r];
                outp[((size_t)(b * Sdim + row)) * Ddim + hh * HD + nt * 16 + ln] = val;
            }
        }
}

// ---------------------------------------------------------------------------
// Launch. d_in order: hidden_states, Wq, bq, Wk, bk, Wv, bv (all fp32).
// d_ws layout (f16):
//   wdq : 3 * D * D                      =  6.3 MB
//   q   : B*H*S*64                       = 18.9 MB
//   k   : B*H*S*64                       = 18.9 MB
//   v   : B*H*64*SPAD (transposed, pad)  = 19.9 MB
// ---------------------------------------------------------------------------
extern "C" void kernel_launch(void* const* d_in, const int* in_sizes, int n_in,
                              void* d_out, int out_size, void* d_ws, size_t ws_size,
                              hipStream_t stream) {
    const float* x  = (const float*)d_in[0];
    const float* Wq = (const float*)d_in[1];
    const float* bq = (const float*)d_in[2];
    const float* Wk = (const float*)d_in[3];
    const float* bk = (const float*)d_in[4];
    const float* Wv = (const float*)d_in[5];
    const float* bv = (const float*)d_in[6];

    _Float16* wdq  = (_Float16*)d_ws;
    _Float16* qbuf = wdq  + (size_t)3 * Ddim * Ddim;
    _Float16* kbuf = qbuf + (size_t)Bdim * Hdim * Sdim * HD;
    _Float16* vbuf = kbuf + (size_t)Bdim * Hdim * Sdim * HD;
    float*    out  = (float*)d_out;

    quant_w<<<dim3(Ddim, 3), 256, 0, stream>>>(Wq, Wk, Wv, wdq);

    qkv_gemm<<<dim3((NTOK + 127) / 128, Ddim / 64, 3), 256, 0, stream>>>(
        x, wdq, bq, bk, bv, qbuf, kbuf, vbuf);

    attn<<<dim3(Bdim * Hdim, (Sdim + 127) / 128), 256, 0, stream>>>(
        qbuf, kbuf, vbuf, out);
}
